// NodeEmbeddingAggregatorGate_87668872446565
// MI455X (gfx1250) — compile-verified, matrix operand order checked by reading, not run on hardware
//
#include <hip/hip_runtime.h>

#define DIM   128
#define TILE  128
#define PADF  136           // 128 + 8 DWORD pad (matches TDM pad_interval=6 / pad_amount=7)
#define NUM_GRAPHS 1024

typedef __attribute__((ext_vector_type(16))) __bf16 v16bf;
typedef __attribute__((ext_vector_type(8)))  float  v8f;
typedef __attribute__((ext_vector_type(4)))  unsigned int u32x4;
typedef __attribute__((ext_vector_type(8)))  int    i32x8;
typedef __attribute__((ext_vector_type(4)))  int    i32x4;

#if __has_builtin(__builtin_amdgcn_tensor_load_to_lds) && \
    __has_builtin(__builtin_amdgcn_s_wait_tensorcnt)
#define USE_TDM 1
#else
#define USE_TDM 0
#endif

__global__ void zero_out_kernel(float* __restrict__ out, int n) {
  int i = blockIdx.x * blockDim.x + threadIdx.x;
  if (i < n) out[i] = 0.0f;
}

__global__ __launch_bounds__(256)
void gate_pool_kernel(const float* __restrict__ x,
                      const int*   __restrict__ bidx,
                      const float* __restrict__ W1,
                      const float* __restrict__ b1,
                      const float* __restrict__ W2,
                      const float* __restrict__ b2,
                      float* __restrict__ out,
                      int N) {
  __shared__ float  xs[TILE][PADF];      // X tile, f32 (reused by scatter phase)
  __shared__ __bf16 w1t[DIM][PADF];      // W1 transposed: w1t[n][k]
  __shared__ float  gates[TILE];
  __shared__ int    bidx_s[TILE];

  const int tid  = threadIdx.x;
  const int lane = tid & 31;
  const int wave = tid >> 5;
  const long row0 = (long)blockIdx.x * TILE;

  // ---- stage W1 (f32 -> bf16, transposed so B-fragment reads are contiguous)
  for (int i = tid; i < DIM * DIM; i += 256) {
    int k = i >> 7, n = i & 127;
    w1t[n][k] = (__bf16)W1[i];
  }

#if USE_TDM
  // ---- stage X tile with the Tensor Data Mover: one instruction moves the
  // whole 128x128 f32 tile global->LDS, inserts the 8-DWORD row padding in
  // hardware, and zero-fills rows past the tensor edge (the N tail).
  if (wave == 0) {
    const unsigned lds_addr = (unsigned)(uintptr_t)&xs[0][0];
    const unsigned long long ga =
        (unsigned long long)(uintptr_t)(x + row0 * DIM);
    long rem = (long)N - row0;                 // valid rows from tile start
    unsigned td1 = (unsigned)(rem > 0 ? rem : 0);

    u32x4 g0;
    g0[0] = 1u;                                  // count=1, user descriptor
    g0[1] = lds_addr;                            // lds_addr [63:32]
    g0[2] = (unsigned)(ga & 0xFFFFFFFFull);      // global_addr [95:64]
    g0[3] = (unsigned)((ga >> 32) & 0x1FFFFFFull) | (2u << 30); // addr hi + type=2

    i32x8 g1;
    g1[0] = (int)((2u << 16)      // data_size = 4B
                | (1u << 20)      // pad_enable
                | (6u << 22)      // pad_interval: every 128 DWORDs
                | (7u << 25));    // pad_amount: 8 DWORDs  -> row stride 136
    g1[1] = (int)(((unsigned)DIM & 0xFFFFu) << 16);        // tensor_dim0 lo16
    g1[2] = (int)((td1 & 0xFFFFu) << 16);                  // dim0 hi=0 | tensor_dim1 lo16
    g1[3] = (int)(((td1 >> 16) & 0xFFFFu) | ((unsigned)TILE << 16)); // dim1 hi | tile_dim0
    g1[4] = TILE;                                          // tile_dim1=128, tile_dim2=0
    g1[5] = DIM;                                           // tensor_dim0_stride lo32
    g1[6] = 0;                                             // stride hi | dim1_stride lo
    g1[7] = 0;

    i32x4 gz = {0, 0, 0, 0};
#if __clang_major__ >= 23
    i32x8 gz8 = {0, 0, 0, 0, 0, 0, 0, 0};
    __builtin_amdgcn_tensor_load_to_lds(g0, g1, gz, gz, gz8, 0);
#else
    __builtin_amdgcn_tensor_load_to_lds(g0, g1, gz, gz, 0);
#endif
    __builtin_amdgcn_s_wait_tensorcnt(0);
  }
#else
  // ---- fallback: manual staging via VGPRs
  {
    const float4* xv = (const float4*)x;
    for (int i = tid; i < TILE * (DIM / 4); i += 256) {
      int r  = i >> 5;
      int c4 = i & 31;
      long row = row0 + r;
      float4 v = make_float4(0.f, 0.f, 0.f, 0.f);
      if (row < N) v = xv[row * (DIM / 4) + c4];
      xs[r][c4 * 4 + 0] = v.x;
      xs[r][c4 * 4 + 1] = v.y;
      xs[r][c4 * 4 + 2] = v.z;
      xs[r][c4 * 4 + 3] = v.w;
    }
  }
#endif

  if (tid < TILE) {
    long row = row0 + tid;
    bidx_s[tid] = (row < N) ? bidx[row] : -1;
  }
  __syncthreads();

  // ---- per-wave GEMM: rows [wave*16, wave*16+16) x all 128 cols of h
  const int nlane = lane & 15;
  const int khalf = lane >> 4;

  float b1r[8], w2r[8];
#pragma unroll
  for (int nt = 0; nt < 8; ++nt) {
    b1r[nt] = b1[nt * 16 + nlane];
    w2r[nt] = W2[nt * 16 + nlane];
  }

  v8f acc[8];
#pragma unroll
  for (int nt = 0; nt < 8; ++nt) {
    v8f z = {};
    acc[nt] = z;
  }

  const int mrow = wave * 16 + nlane;   // A-matrix row this lane supplies
#pragma unroll
  for (int kb = 0; kb < 4; ++kb) {
    // A fragment (16-bit A 16x32 layout): lane<16 -> K {0..7,16..23}+kb*32,
    // lane>=16 -> K {8..15,24..31}+kb*32
    const float* ap = &xs[mrow][kb * 32 + khalf * 8];
    v16bf afrag;
#pragma unroll
    for (int e = 0; e < 8; ++e) {
      afrag[e]     = (__bf16)ap[e];
      afrag[e + 8] = (__bf16)ap[16 + e];
    }
#pragma unroll
    for (int nt = 0; nt < 8; ++nt) {
      const __bf16* bp = &w1t[nt * 16 + nlane][kb * 32 + khalf * 8];
      v16bf bfrag;
#pragma unroll
      for (int e = 0; e < 8; ++e) {
        bfrag[e]     = bp[e];
        bfrag[e + 8] = bp[16 + e];
      }
      acc[nt] = __builtin_amdgcn_wmma_f32_16x16x32_bf16(
          false, afrag, false, bfrag, (short)0, acc[nt], false, false);
    }
  }

  // ---- bias + relu + dot(h_row, W2), cross-lane reduce, sigmoid -> gates
  const float bias2 = b2[0];
#pragma unroll
  for (int r = 0; r < 8; ++r) {
    float p = 0.f;
#pragma unroll
    for (int nt = 0; nt < 8; ++nt) {
      float h = acc[nt][r] + b1r[nt];
      h = fmaxf(h, 0.f);
      p += h * w2r[nt];
    }
    // reduce over the 16 lanes of each half (bits 0..3 never cross bit 4)
    p += __shfl_xor(p, 1, 32);
    p += __shfl_xor(p, 2, 32);
    p += __shfl_xor(p, 4, 32);
    p += __shfl_xor(p, 8, 32);
    float g = 1.f / (1.f + __expf(-(p + bias2)));
    // C/D layout: lanes 0-15 hold M=r, lanes 16-31 hold M=r+8
    if (nlane == 0) gates[wave * 16 + khalf * 8 + r] = g;
  }
  __syncthreads();

  // ---- gated segment-sum: 2 threads per column walk 64 sorted rows each,
  // flush to L2-resident output on segment change
  const int d    = tid & 127;
  const int half = tid >> 7;
  const int rbeg = half * 64;
  float accv = 0.f;
  int cur = -1;
  for (int r = rbeg; r < rbeg + 64; ++r) {
    long row = row0 + r;
    if (row >= N) break;            // batch_idx is sorted; tail rows only
    int idx = bidx_s[r];
    if (idx != cur) {
      if (cur >= 0)
        __hip_atomic_fetch_add(&out[cur * DIM + d], accv,
                               __ATOMIC_RELAXED, __HIP_MEMORY_SCOPE_AGENT);
      cur = idx;
      accv = 0.f;
    }
    accv += gates[r] * xs[r][d];
  }
  if (cur >= 0)
    __hip_atomic_fetch_add(&out[cur * DIM + d], accv,
                           __ATOMIC_RELAXED, __HIP_MEMORY_SCOPE_AGENT);
}

extern "C" void kernel_launch(void* const* d_in, const int* in_sizes, int n_in,
                              void* d_out, int out_size, void* d_ws, size_t ws_size,
                              hipStream_t stream) {
  const float* x   = (const float*)d_in[0];
  const int*   bi  = (const int*)d_in[1];
  const float* W1  = (const float*)d_in[2];
  const float* b1  = (const float*)d_in[3];
  const float* W2  = (const float*)d_in[4];
  const float* b2  = (const float*)d_in[5];
  float*       out = (float*)d_out;

  const int N = in_sizes[0] / DIM;

  // zero the (poisoned) output before atomic accumulation
  zero_out_kernel<<<(out_size + 255) / 256, 256, 0, stream>>>(out, out_size);

  const int tiles = (N + TILE - 1) / TILE;
  gate_pool_kernel<<<tiles, 256, 0, stream>>>(x, bi, W1, b1, W2, b2, out, N);
}
